// KANSplineLayer_75290776699185
// MI455X (gfx1250) — matrix operand order repaired and checked by reading, not compiled, not run on hardware
//
#include <hip/hip_runtime.h>
#include <hip/hip_bf16.h>

// ---------------------------------------------------------------------------
// KAN spline layer, MI455X (gfx1250, wave32, WMMA)
//   out = x @ W1^T + exp(-(x-mean(grid))^2) @ sum_g(grid)
//
// Prep passes (one-shot, streaming):
//   P1 grid(128MB) -> W2 bf16 [k][n] + global sum (mean)
//   P2 base_weight -> bf16 [n][k]
//   P3 x -> bf16 x, bf16 basis=exp(-(x-mu)^2)   (exp computed once)
//   P4 W2 [k][n] -> W2^T bf16 [n][k] (tiled LDS transpose)
// GEMM pass: pure-bf16 dual GEMM, all operands [row][K]; working set 80MB
// (fits 192MB L2). Inner loop: GLOBAL_LOAD_ASYNC_TO_LDS_B128 staging +
// ds_load_b128 fragments + 16x v_wmma_f32_16x16x32_bf16 per wave per K-step.
// ---------------------------------------------------------------------------

typedef __bf16 bf16_t;
typedef bf16_t v16bf __attribute__((ext_vector_type(16)));
typedef bf16_t v8bf  __attribute__((ext_vector_type(8)));
typedef float  v8f   __attribute__((ext_vector_type(8)));
typedef float  v4f   __attribute__((ext_vector_type(4)));

#define IN_F   2048
#define OUT_F  2048
#define GDIM   8
#define BATCH  8192

#define BM 128
#define BN 128
#define BK 32
#define LDST 48   // bf16 elements per LDS row (32 data + 16 pad) -> 96B rows

// Async global->LDS staging via inline asm (clang builtin has unspellable
// pointer types on this toolchain). Flip to 0 to use the VGPR-copy fallback.
#define USE_ASYNC_LDS 1

static __device__ __forceinline__ v8bf cvt8(const v4f& a, const v4f& b) {
  v8bf r;
  r[0] = (bf16_t)a[0]; r[1] = (bf16_t)a[1];
  r[2] = (bf16_t)a[2]; r[3] = (bf16_t)a[3];
  r[4] = (bf16_t)b[0]; r[5] = (bf16_t)b[1];
  r[6] = (bf16_t)b[2]; r[7] = (bf16_t)b[3];
  return r;
}

// ---- P1: grid -> W2 bf16 [k][n]; ws[0] += sum(grid) -------------------------
__global__ void kan_prep(const float* __restrict__ grid,
                         bf16_t* __restrict__ w2kn,
                         float* __restrict__ mean_sum) {
  __shared__ float red[256];
  const int idx = blockIdx.x * 256 + threadIdx.x;      // < IN_F*OUT_F
  const v4f* g4 = (const v4f*)grid;
  v4f a = g4[(size_t)idx * 2 + 0];
  v4f b = g4[(size_t)idx * 2 + 1];
  float s = (a[0] + a[1]) + (a[2] + a[3]) + (b[0] + b[1]) + (b[2] + b[3]);
  w2kn[idx] = (bf16_t)s;
  red[threadIdx.x] = s;
  __syncthreads();
#pragma unroll
  for (int off = 128; off > 0; off >>= 1) {
    if (threadIdx.x < off) red[threadIdx.x] += red[threadIdx.x + off];
    __syncthreads();
  }
  if (threadIdx.x == 0) atomicAdd(mean_sum, red[0]);
}

// ---- P2: base_weight f32 -> bf16 (layout already [n][k]) --------------------
__global__ void kan_cvt_w1(const float* __restrict__ w1, bf16_t* __restrict__ w1b) {
  const size_t i8 = ((size_t)blockIdx.x * 256 + threadIdx.x) * 8;
  const v4f* p = (const v4f*)(w1 + i8);
  *(v8bf*)(w1b + i8) = cvt8(p[0], p[1]);
}

// ---- P3: x -> bf16 x, bf16 basis=exp(-(x-mu)^2) -----------------------------
__global__ void kan_cvt_x(const float* __restrict__ x,
                          const float* __restrict__ mean_sum,
                          bf16_t* __restrict__ xb, bf16_t* __restrict__ bb) {
  const float mu = mean_sum[0] * (1.0f / 33554432.0f);   // / (2048*2048*8)
  const size_t i8 = ((size_t)blockIdx.x * 256 + threadIdx.x) * 8;
  const v4f* p = (const v4f*)(x + i8);
  v4f a = p[0], b = p[1], ea, eb;
#pragma unroll
  for (int e = 0; e < 4; ++e) {
    float da = a[e] - mu; ea[e] = __expf(-da * da);
    float db = b[e] - mu; eb[e] = __expf(-db * db);
  }
  *(v8bf*)(xb + i8) = cvt8(a, b);
  *(v8bf*)(bb + i8) = cvt8(ea, eb);
}

// ---- P4: W2 bf16 [k][n] -> [n][k] (64x64 LDS tile transpose) ----------------
__global__ void kan_tr_w2(const bf16_t* __restrict__ src, bf16_t* __restrict__ dst) {
  __shared__ bf16_t tile[64][68];
  const int K0 = blockIdx.x * 64;
  const int N0 = blockIdx.y * 64;
  const int t = threadIdx.x;
#pragma unroll
  for (int e = 0; e < 16; ++e) {
    const int id = t + e * 256;
    const int k = id >> 6, n = id & 63;
    tile[k][n] = src[(size_t)(K0 + k) * OUT_F + N0 + n];
  }
  __syncthreads();
#pragma unroll
  for (int e = 0; e < 16; ++e) {
    const int id = t + e * 256;
    const int n = id >> 6, k = id & 63;
    dst[(size_t)(N0 + n) * IN_F + K0 + k] = tile[k][n];
  }
}

// ---- GEMM: out = xb @ w1b^T(K-major) + bb @ w2t^T(K-major), f32 accum -------
__global__ __launch_bounds__(256)
void kan_gemm(const bf16_t* __restrict__ xb,   // (BATCH, IN_F)
              const bf16_t* __restrict__ bb,   // (BATCH, IN_F)
              const bf16_t* __restrict__ w1b,  // (OUT_F, IN_F)
              const bf16_t* __restrict__ w2t,  // (OUT_F, IN_F)
              float* __restrict__ out) {
  // double-buffered stages: 2 * 4 * 128 * 48 * 2B = 96 KB
  __shared__ bf16_t Ax[2][BM * LDST];
  __shared__ bf16_t Ab[2][BM * LDST];
  __shared__ bf16_t B1[2][BN * LDST];
  __shared__ bf16_t B2[2][BN * LDST];

  const int tid = threadIdx.x;
  const int M0 = blockIdx.y * BM;
  const int N0 = blockIdx.x * BN;

  // ---- staging mapping: 256 threads, 16 bf16 (2x16B) per array each ----
  const int rowS = tid >> 1;            // 0..127
  const int hS   = (tid & 1) * 16;      // k offset 0/16
  const bf16_t* gx = xb  + (size_t)(M0 + rowS) * IN_F + hS;
  const bf16_t* gb = bb  + (size_t)(M0 + rowS) * IN_F + hS;
  const bf16_t* g1 = w1b + (size_t)(N0 + rowS) * IN_F + hS;
  const bf16_t* g2 = w2t + (size_t)(N0 + rowS) * IN_F + hS;
  const int ldsOff = rowS * LDST + hS;

#if USE_ASYNC_LDS
  // GLOBAL_LOAD_ASYNC_TO_LDS_B128: vdst = LDS byte offset (flat addr[31:0]),
  // vaddr = 64-bit global address. IOFFSET is added to BOTH the global and
  // the LDS address (ISA 08_async_tensor 4.4), so the second 16B chunk is
  // issued as offset:16 from the same address registers. Tracked by ASYNCcnt.
  auto cp32B = [&](const bf16_t* g, bf16_t* l) {
    const unsigned lds = (unsigned)(unsigned long long)l;
    const unsigned long long ga = (unsigned long long)g;
    asm volatile("global_load_async_to_lds_b128 %0, %1, off"
                 :: "v"(lds), "v"(ga) : "memory");
    asm volatile("global_load_async_to_lds_b128 %0, %1, off offset:16"
                 :: "v"(lds), "v"(ga) : "memory");
  };
  auto issue = [&](int buf, int k0) {
    cp32B(gx + k0, &Ax[buf][ldsOff]);
    cp32B(gb + k0, &Ab[buf][ldsOff]);
    cp32B(g1 + k0, &B1[buf][ldsOff]);
    cp32B(g2 + k0, &B2[buf][ldsOff]);
  };
  auto wait_async = [&]() {
    asm volatile("s_wait_asynccnt 0" ::: "memory");
  };
#else
  v8bf rg[8];
  auto fetch = [&](int k0) {
    rg[0] = *(const v8bf*)(gx + k0); rg[1] = *(const v8bf*)(gx + k0 + 8);
    rg[2] = *(const v8bf*)(gb + k0); rg[3] = *(const v8bf*)(gb + k0 + 8);
    rg[4] = *(const v8bf*)(g1 + k0); rg[5] = *(const v8bf*)(g1 + k0 + 8);
    rg[6] = *(const v8bf*)(g2 + k0); rg[7] = *(const v8bf*)(g2 + k0 + 8);
  };
  auto stage = [&](int buf) {
    *(v8bf*)&Ax[buf][ldsOff] = rg[0]; *(v8bf*)&Ax[buf][ldsOff + 8] = rg[1];
    *(v8bf*)&Ab[buf][ldsOff] = rg[2]; *(v8bf*)&Ab[buf][ldsOff + 8] = rg[3];
    *(v8bf*)&B1[buf][ldsOff] = rg[4]; *(v8bf*)&B1[buf][ldsOff + 8] = rg[5];
    *(v8bf*)&B2[buf][ldsOff] = rg[6]; *(v8bf*)&B2[buf][ldsOff + 8] = rg[7];
  };
#endif

  // ---- per-wave compute mapping ----
  const int lane = tid & 31;
  const int l16  = lane & 15;
  const int hi8  = (lane >> 4) * 8;     // A frag: K {0-7,16-23} vs {8-15,24-31}
  const int hi16 = (lane >> 4) * 8;     // B frag: K {0-7,16-23} vs {8-15,24-31}
  const int wave = tid >> 5;
  const int wm = (wave & 1) * 64;       // 2 waves along M, wave tile 64x32
  const int wn = (wave >> 1) * 32;      // 4 waves along N

  auto loadA = [&](const bf16_t* base) -> v16bf {
    v8bf lo = *(const v8bf*)(base);
    v8bf hi = *(const v8bf*)(base + 16);
    return __builtin_shufflevector(lo, hi, 0,1,2,3,4,5,6,7,8,9,10,11,12,13,14,15);
  };
  auto loadB = [&](const bf16_t* base) -> v16bf {
    v8bf lo = *(const v8bf*)(base);
    v8bf hi = *(const v8bf*)(base + 16);
    return __builtin_shufflevector(lo, hi, 0,1,2,3,4,5,6,7,8,9,10,11,12,13,14,15);
  };

  v8f acc[4][2] = {};

  auto compute = [&](int buf) {
    v16bf a1[4], a2[4], bb1[2], bb2[2];
#pragma unroll
    for (int mi = 0; mi < 4; ++mi) {
      const int row = wm + mi * 16 + l16;
      a1[mi] = loadA(&Ax[buf][row * LDST + hi8]);
      a2[mi] = loadA(&Ab[buf][row * LDST + hi8]);
    }
#pragma unroll
    for (int ni = 0; ni < 2; ++ni) {
      const int col = wn + ni * 16 + l16;
      bb1[ni] = loadB(&B1[buf][col * LDST + hi16]);
      bb2[ni] = loadB(&B2[buf][col * LDST + hi16]);
    }
#pragma unroll
    for (int mi = 0; mi < 4; ++mi) {
#pragma unroll
      for (int ni = 0; ni < 2; ++ni) {
        acc[mi][ni] = __builtin_amdgcn_wmma_f32_16x16x32_bf16(
            false, a1[mi], false, bb1[ni], (short)0, acc[mi][ni], false, false);
        acc[mi][ni] = __builtin_amdgcn_wmma_f32_16x16x32_bf16(
            false, a2[mi], false, bb2[ni], (short)0, acc[mi][ni], false, false);
      }
    }
  };

  // ---- pipelined K loop, one barrier per iteration ----
  const int STEPS = IN_F / BK;  // 64
#if USE_ASYNC_LDS
  issue(0, 0);
  wait_async();
  __syncthreads();
  for (int t = 0; t < STEPS; ++t) {
    const bool more = (t + 1) < STEPS;
    if (more) issue((t + 1) & 1, (t + 1) * BK);  // DMA next stage over compute
    compute(t & 1);
    if (more) wait_async();                      // own async copies done
    __syncthreads();                             // everyone's copies done
  }
#else
  fetch(0);
  stage(0);
  __syncthreads();
  for (int t = 0; t < STEPS; ++t) {
    const bool more = (t + 1) < STEPS;
    if (more) fetch((t + 1) * BK);               // loads in flight over compute
    compute(t & 1);
    if (more) stage((t + 1) & 1);
    __syncthreads();
  }
#endif

  // ---- epilogue: C/D layout (M = vgpr + 8*(lane>=16), N = lane&15) ----
  const int mh = (lane >> 4) * 8;
#pragma unroll
  for (int mi = 0; mi < 4; ++mi) {
#pragma unroll
    for (int ni = 0; ni < 2; ++ni) {
      const int col = N0 + wn + ni * 16 + l16;
#pragma unroll
      for (int v = 0; v < 8; ++v) {
        const int row = M0 + wm + mi * 16 + v + mh;
        out[(size_t)row * OUT_F + col] = acc[mi][ni][v];
      }
    }
  }
}

// ---------------------------------------------------------------------------
extern "C" void kernel_launch(void* const* d_in, const int* in_sizes, int n_in,
                              void* d_out, int out_size, void* d_ws, size_t ws_size,
                              hipStream_t stream) {
  (void)in_sizes; (void)n_in; (void)out_size; (void)ws_size;
  const float* x    = (const float*)d_in[0];
  const float* bw   = (const float*)d_in[1];   // (OUT_F, IN_F)
  const float* grid = (const float*)d_in[2];   // (IN_F, OUT_F, GDIM)
  float* out = (float*)d_out;

  // workspace layout (bytes)
  char* ws = (char*)d_ws;
  float*  mean_sum = (float*)ws;                                   // 256 B
  bf16_t* w2kn = (bf16_t*)(ws + 256);                              // 8 MB
  bf16_t* w2t  = (bf16_t*)(ws + 256 + (8u << 20));                 // 8 MB
  bf16_t* w1b  = (bf16_t*)(ws + 256 + (16u << 20));                // 8 MB
  bf16_t* xb   = (bf16_t*)(ws + 256 + (24u << 20));                // 32 MB
  bf16_t* bb   = (bf16_t*)(ws + 256 + (56u << 20));                // 32 MB

  (void)hipMemsetAsync(d_ws, 0, 256, stream);

  kan_prep<<<(IN_F * OUT_F) / 256, 256, 0, stream>>>(grid, w2kn, mean_sum);
  kan_cvt_w1<<<(OUT_F * IN_F) / (256 * 8), 256, 0, stream>>>(bw, w1b);
  kan_cvt_x<<<(BATCH * IN_F) / (256 * 8), 256, 0, stream>>>(x, mean_sum, xb, bb);
  {
    dim3 g(IN_F / 64, OUT_F / 64);
    kan_tr_w2<<<g, 256, 0, stream>>>(w2kn, w2t);
  }
  {
    dim3 g(OUT_F / BN, BATCH / BM);
    kan_gemm<<<g, 256, 0, stream>>>(xb, bb, w1b, w2t, out);
  }
}
